// CROSSTRANS_81449759801700
// MI455X (gfx1250) — matrix operand order repaired and checked by reading, not compiled, not run on hardware
//
#include <hip/hip_runtime.h>
#include <stdint.h>

typedef unsigned short u16;
typedef __attribute__((ext_vector_type(16))) __bf16 v16bf;
typedef __attribute__((ext_vector_type(8)))  float  v8f;

struct Bf16Tile { uint4 lo, hi; };

__device__ __forceinline__ u16 f2bf(float x) {
    unsigned u = __float_as_uint(x);
    unsigned r = (u + 0x7FFFu + ((u >> 16) & 1u)) >> 16;   // RNE
    return (u16)r;
}
__device__ __forceinline__ float hswish(float x) {
    return x * fminf(fmaxf(x + 3.0f, 0.0f), 6.0f) * (1.0f / 6.0f);
}

// A-matrix 16x32 bf16 per CDNA5 layout: lane = (m & 15) | (half<<4).
// lane holds K = [8*half, 8*half+8) and [16+8*half, 16+8*half+8)  (two 16B chunks)
__device__ __forceinline__ v16bf load_a16(const u16* A, int ldk, int m, int k0, int half) {
    const u16* p = A + (size_t)m * ldk + k0 + 8 * half;
    Bf16Tile t;
    t.lo = *(const uint4*)p;
    t.hi = *(const uint4*)(p + 16);
    return __builtin_bit_cast(v16bf, t);
}
// B-matrix 32x16 bf16: lane holds column n = lane&15, K = [8*half, 8*half+8) and [16+8*half, 16+8*half+8) (two 16B chunks)
__device__ __forceinline__ v16bf load_b16(const u16* Bm, int ldk, int n, int k0, int half) {
    const u16* p = Bm + (size_t)n * ldk + k0 + 8 * half;
    Bf16Tile t;
    t.lo = *(const uint4*)p;
    t.hi = *(const uint4*)(p + 16);
    return __builtin_bit_cast(v16bf, t);
}
__device__ __forceinline__ v16bf load_a_row(const u16* Arow, int k0, int half) {
    const u16* p = Arow + k0 + 8 * half;
    Bf16Tile t;
    t.lo = *(const uint4*)p;
    t.hi = *(const uint4*)(p + 16);
    return __builtin_bit_cast(v16bf, t);
}
__device__ __forceinline__ v16bf load_b_row(const u16* Brow, int k0, int half) {
    const u16* p = Brow + k0 + 8 * half;
    Bf16Tile t;
    t.lo = *(const uint4*)p;
    t.hi = *(const uint4*)(p + 16);
    return __builtin_bit_cast(v16bf, t);
}
__device__ __forceinline__ v8f wmma_bf16(v16bf a, v16bf b, v8f c) {
    return __builtin_amdgcn_wmma_f32_16x16x32_bf16(false, a, false, b, (short)0, c, false, false);
}

// ---------------------------------------------------------------------------
// Generic GEMM: C[M,N](f32) = A[M,K](bf16, row-major) x W[N,K](bf16, row-major)^T
// Block: 256 thr = 8 waves; wave -> 16(M) x 64(N); block tile 128(M) x 64(N).
// Register double-buffered K pipeline + global_prefetch; NT stores for C.
// ---------------------------------------------------------------------------
__global__ __launch_bounds__(256) void gemm_bf16_nt(
    const u16* __restrict__ A, const u16* __restrict__ W, float* __restrict__ C,
    int M, int N, int K)
{
    const int lane = threadIdx.x & 31;
    const int wave = threadIdx.x >> 5;
    const int col  = lane & 15;
    const int half = lane >> 4;
    const int n0 = blockIdx.x * 64;
    const int m0 = blockIdx.y * 128 + wave * 16;
    int mload = m0 + col; if (mload >= M) mload = M - 1;

    const u16* Arow = A + (size_t)mload * K;
    const u16* W0 = W + (size_t)(n0      + col) * K;
    const u16* W1 = W + (size_t)(n0 + 16 + col) * K;
    const u16* W2 = W + (size_t)(n0 + 32 + col) * K;
    const u16* W3 = W + (size_t)(n0 + 48 + col) * K;

    v8f acc[4] = {};
    // prologue: tiles for k0 = 0
    v16bf a  = load_a_row(Arow, 0, half);
    v16bf b0 = load_b_row(W0, 0, half);
    v16bf b1 = load_b_row(W1, 0, half);
    v16bf b2 = load_b_row(W2, 0, half);
    v16bf b3 = load_b_row(W3, 0, half);

    for (int k0 = 0; k0 < K; k0 += 32) {
        const int kn = (k0 + 32 < K) ? k0 + 32 : k0;     // branchless tail reload
        // issue next-iteration loads before this iteration's WMMAs
        v16bf an = load_a_row(Arow, kn, half);
        v16bf c0 = load_b_row(W0, kn, half);
        v16bf c1 = load_b_row(W1, kn, half);
        v16bf c2 = load_b_row(W2, kn, half);
        v16bf c3 = load_b_row(W3, kn, half);
        if (k0 + 256 < K) {                               // global_prefetch_b8, ~8 iters ahead
            __builtin_prefetch(Arow + k0 + 256, 0, 1);
            __builtin_prefetch(W0 + k0 + 256, 0, 1);
            __builtin_prefetch(W1 + k0 + 256, 0, 1);
            __builtin_prefetch(W2 + k0 + 256, 0, 1);
            __builtin_prefetch(W3 + k0 + 256, 0, 1);
        }
        acc[0] = wmma_bf16(a, b0, acc[0]);
        acc[1] = wmma_bf16(a, b1, acc[1]);
        acc[2] = wmma_bf16(a, b2, acc[2]);
        acc[3] = wmma_bf16(a, b3, acc[3]);
        a = an; b0 = c0; b1 = c1; b2 = c2; b3 = c3;
    }
#pragma unroll
    for (int j = 0; j < 4; ++j) {
        int n = n0 + j * 16 + col;
#pragma unroll
        for (int r = 0; r < 8; ++r) {
            int m = m0 + 8 * half + r;     // D layout: VGPR r -> M = r + 8*half
            if (m < M) __builtin_nontemporal_store(acc[j][r], &C[(size_t)m * N + n]);
        }
    }
}

// ---------------------------------------------------------------------------
// Elementwise / transforms
// ---------------------------------------------------------------------------
__global__ __launch_bounds__(256) void to_bf16_k(const float* __restrict__ in,
                                                 u16* __restrict__ out, size_t n) {
    size_t i = (size_t)blockIdx.x * 256 + threadIdx.x;
    if (i < n) out[i] = f2bf(in[i]);
}

// im_emd (B,C,H,W) f32 -> xbf (B, N=H*W, C) bf16 via LDS tile transpose
__global__ __launch_bounds__(256) void transpose_to_bf16(const float* __restrict__ im,
                                                         u16* __restrict__ xbf) {
    __shared__ float t[32][33];
    const int b  = blockIdx.z;
    const int n0 = blockIdx.x * 32;
    const int c0 = blockIdx.y * 32;
    const int tx = threadIdx.x & 31;
    const int ty = threadIdx.x >> 5;          // 0..7
    const float* src = im + (size_t)b * 1024 * 9216;
#pragma unroll
    for (int j = 0; j < 32; j += 8)
        t[ty + j][tx] = src[(size_t)(c0 + ty + j) * 9216 + n0 + tx];
    __syncthreads();
    u16* dst = xbf + (size_t)b * 9216 * 1024;
#pragma unroll
    for (int j = 0; j < 32; j += 8)
        dst[(size_t)(n0 + ty + j) * 1024 + c0 + tx] = f2bf(t[tx][ty + j]);
}

// column sum / sumsq with row-validity mask (row % period < valid); NT reads
__global__ __launch_bounds__(256) void colstats(const float* __restrict__ Y,
                                                float* __restrict__ sum, float* __restrict__ ssq,
                                                int R, int Cc, int period, int valid, int rpc) {
    int c = blockIdx.x * 256 + threadIdx.x;
    if (c >= Cc) return;
    int r0 = blockIdx.y * rpc;
    int r1 = r0 + rpc; if (r1 > R) r1 = R;
    float s = 0.f, q = 0.f;
    for (int r = r0; r < r1; ++r) {
        if ((r % period) < valid) {
            float v = __builtin_nontemporal_load(&Y[(size_t)r * Cc + c]);
            s += v; q += v * v;
        }
    }
    atomicAdd(&sum[c], s);
    atomicAdd(&ssq[c], q);
}

__global__ __launch_bounds__(256) void make_scaleshift(const float* __restrict__ sum,
                                                       const float* __restrict__ ssq,
                                                       const float* __restrict__ g,
                                                       const float* __restrict__ b,
                                                       float* __restrict__ scl, float* __restrict__ shf,
                                                       int Cc, float invR) {
    int c = blockIdx.x * 256 + threadIdx.x;
    if (c >= Cc) return;
    float mu  = sum[c] * invR;
    float var = ssq[c] * invR - mu * mu;
    float s   = g[c] * rsqrtf(var + 1e-5f);
    scl[c] = s;
    shf[c] = b[c] - mu * s;
}

__global__ __launch_bounds__(256) void bn_hswish_bf16(const float* __restrict__ y,
                                                      const float* __restrict__ scl,
                                                      const float* __restrict__ shf,
                                                      u16* __restrict__ o, size_t total, int Cc) {
    size_t i = (size_t)blockIdx.x * 256 + threadIdx.x;
    if (i >= total) return;
    int c = (int)(i % Cc);
    float v = __builtin_nontemporal_load(&y[i]);
    o[i] = f2bf(hswish(v * scl[c] + shf[c]));
}

// y2 (B*N, 1536) -> kbuf (B,NH,N,KD) bf16 and vT (B,NH,D_V,N) bf16
__global__ __launch_bounds__(256) void scatter_kv(const float* __restrict__ y2,
                                                  const float* __restrict__ scl,
                                                  const float* __restrict__ shf,
                                                  u16* __restrict__ kbuf, u16* __restrict__ vT) {
    size_t i = (size_t)blockIdx.x * 256 + threadIdx.x;
    if (i >= (size_t)73728 * 1536) return;
    int c = (int)(i % 1536);
    size_t row = i / 1536;
    int b = (int)(row / 9216);
    int n = (int)(row % 9216);
    float v = __builtin_nontemporal_load(&y2[i]);
    u16 bv = f2bf(v * scl[c] + shf[c]);
    int h = c / 192, r = c % 192;
    if (r < 64) kbuf[((size_t)(b * 8 + h) * 9216 + n) * 64 + r] = bv;
    else        vT[((size_t)(b * 8 + h) * 128 + (r - 64)) * 9216 + n] = bv;
}

// tiny q GEMM (60x512x512), f32
__global__ __launch_bounds__(512) void q_gemm(const float* __restrict__ text,
                                              const float* __restrict__ Wq,
                                              float* __restrict__ yq) {
    int t = blockIdx.x, c = threadIdx.x;
    float s = 0.f;
    for (int k = 0; k < 512; ++k) s += text[t * 512 + k] * Wq[c * 512 + k];
    yq[t * 512 + c] = s;
}

// BN over 60 rows + write padded (64x512) bf16 A-matrix, rows 60..63 zero
__global__ __launch_bounds__(256) void q_bn_fill(const float* __restrict__ yq,
                                                 const float* __restrict__ gq,
                                                 const float* __restrict__ bq,
                                                 u16* __restrict__ qb) {
    int c = blockIdx.x * 256 + threadIdx.x;
    if (c >= 512) return;
    float s = 0.f, q = 0.f;
    for (int t = 0; t < 60; ++t) { float v = yq[t * 512 + c]; s += v; q += v * v; }
    float mu = s / 60.f, var = q / 60.f - mu * mu;
    float sc = gq[c] * rsqrtf(var + 1e-5f);
    float sh = bq[c] - mu * sc;
    for (int t = 0; t < 64; ++t)
        qb[t * 512 + c] = (t < 60) ? f2bf(yq[t * 512 + c] * sc + sh) : (u16)0;
}

// logits[bh][t][n] = (q_h . k_bh,n) * 0.125 + biases[h][100*(n/96) + |t - n%96|]
__global__ __launch_bounds__(128) void attn_logits(const u16* __restrict__ qb,
                                                   const u16* __restrict__ kbuf,
                                                   const float* __restrict__ biases,
                                                   float* __restrict__ logits) {
    const int bh = blockIdx.y;                   // b*8 + h
    const int h  = bh & 7;
    const int n0 = blockIdx.x * 16;
    const int lane = threadIdx.x & 31;
    const int wave = threadIdx.x >> 5;           // m-tile (t block)
    const int col = lane & 15, half = lane >> 4;
    const int m0 = wave * 16;
    const u16* kmat = kbuf + (size_t)bh * 9216 * 64;

    // K=64 -> fully unrolled, all tiles loaded up front (single wait window)
    v16bf a0 = load_a16(qb, 512, m0 + col, h * 64,      half);
    v16bf b0 = load_b16(kmat, 64, n0 + col, 0,          half);
    v16bf a1 = load_a16(qb, 512, m0 + col, h * 64 + 32, half);
    v16bf b1 = load_b16(kmat, 64, n0 + col, 32,         half);
    v8f acc = {};
    acc = wmma_bf16(a0, b0, acc);
    acc = wmma_bf16(a1, b1, acc);

    const int n = n0 + col;
    const int dx = n / 96, ny = n % 96;
    const float* bh_bias = biases + (size_t)h * 10000 + 100 * dx;
#pragma unroll
    for (int r = 0; r < 8; ++r) {
        int t  = m0 + 8 * half + r;
        int dy = t > ny ? t - ny : ny - t;
        __builtin_nontemporal_store(acc[r] * 0.125f + bh_bias[dy],
                                    &logits[((size_t)bh * 64 + t) * 9216 + n]);
    }
}

__global__ __launch_bounds__(256) void softmax_row(const float* __restrict__ logits,
                                                   u16* __restrict__ probs) {
    __shared__ float red[256];
    const int row = blockIdx.x;
    const int t   = row & 63;
    const int tid = threadIdx.x;
    const float* L = logits + (size_t)row * 9216;
    u16* P = probs + (size_t)row * 9216;
    if (t >= 60) {                                 // padded rows: clean zeros
        for (int i = tid; i < 9216; i += 256) P[i] = 0;
        return;
    }
    float m = -3.4e38f;
    for (int i = tid; i < 9216; i += 256) m = fmaxf(m, L[i]);
    red[tid] = m; __syncthreads();
    for (int s = 128; s > 0; s >>= 1) { if (tid < s) red[tid] = fmaxf(red[tid], red[tid + s]); __syncthreads(); }
    const float mx = red[0]; __syncthreads();
    float sum = 0.f;
    for (int i = tid; i < 9216; i += 256) sum += __expf(L[i] - mx);
    red[tid] = sum; __syncthreads();
    for (int s = 128; s > 0; s >>= 1) { if (tid < s) red[tid] += red[tid + s]; __syncthreads(); }
    const float inv = 1.0f / red[0];
    for (int i = tid; i < 9216; i += 256) P[i] = f2bf(__expf(L[i] - mx) * inv);
}

// out[b][t][h][d] += probs[bh] (64x9216) x vT[bh] (128x9216)^T, K split over grid.y
// Register double-buffered pipeline (B tile shared across the 4 M-tiles).
__global__ __launch_bounds__(256) void attn_out(const u16* __restrict__ probs,
                                                const u16* __restrict__ vT,
                                                float* __restrict__ outb) {
    const int bh = blockIdx.x;
    const int b = bh >> 3, h = bh & 7;
    const int ksplit = blockIdx.y;                   // 4 splits of 2304
    const int lane = threadIdx.x & 31;
    const int wave = threadIdx.x >> 5;               // d-tile 0..7
    const int col = lane & 15, half = lane >> 4;
    const int d0 = wave * 16;
    const u16* A0 = probs + ((size_t)bh * 64 + 0  + col) * 9216;
    const u16* A1 = probs + ((size_t)bh * 64 + 16 + col) * 9216;
    const u16* A2 = probs + ((size_t)bh * 64 + 32 + col) * 9216;
    const u16* A3 = probs + ((size_t)bh * 64 + 48 + col) * 9216;
    const u16* Bv = vT + ((size_t)bh * 128 + d0 + col) * 9216;

    v8f acc[4] = {};
    const int kbeg = ksplit * 2304, kend = kbeg + 2304;

    v16bf bm = load_b_row(Bv, kbeg, half);
    v16bf a0 = load_a_row(A0, kbeg, half);
    v16bf a1 = load_a_row(A1, kbeg, half);
    v16bf a2 = load_a_row(A2, kbeg, half);
    v16bf a3 = load_a_row(A3, kbeg, half);
    for (int k0 = kbeg; k0 < kend; k0 += 32) {
        const int kn = (k0 + 32 < kend) ? k0 + 32 : k0;
        v16bf bn = load_b_row(Bv, kn, half);
        v16bf c0 = load_a_row(A0, kn, half);
        v16bf c1 = load_a_row(A1, kn, half);
        v16bf c2 = load_a_row(A2, kn, half);
        v16bf c3 = load_a_row(A3, kn, half);
        if (k0 + 256 < kend) {
            __builtin_prefetch(Bv + k0 + 256, 0, 1);
            __builtin_prefetch(A0 + k0 + 256, 0, 1);
            __builtin_prefetch(A1 + k0 + 256, 0, 1);
            __builtin_prefetch(A2 + k0 + 256, 0, 1);
            __builtin_prefetch(A3 + k0 + 256, 0, 1);
        }
        acc[0] = wmma_bf16(a0, bm, acc[0]);
        acc[1] = wmma_bf16(a1, bm, acc[1]);
        acc[2] = wmma_bf16(a2, bm, acc[2]);
        acc[3] = wmma_bf16(a3, bm, acc[3]);
        bm = bn; a0 = c0; a1 = c1; a2 = c2; a3 = c3;
    }
#pragma unroll
    for (int i = 0; i < 4; ++i) {
#pragma unroll
        for (int r = 0; r < 8; ++r) {
            int t = i * 16 + 8 * half + r;
            atomicAdd(&outb[((size_t)(b * 64 + t) * 8 + h) * 128 + d0 + col], acc[i][r]);
        }
    }
}

__global__ __launch_bounds__(256) void hswish_to_bf16(const float* __restrict__ in,
                                                      u16* __restrict__ out, size_t n) {
    size_t i = (size_t)blockIdx.x * 256 + threadIdx.x;
    if (i < n) out[i] = f2bf(hswish(in[i]));
}

__global__ __launch_bounds__(256) void apply_proj_k(const float* __restrict__ y,
                                                    const float* __restrict__ scl,
                                                    const float* __restrict__ shf,
                                                    const float* __restrict__ text,
                                                    float* __restrict__ tdf, u16* __restrict__ tdbf) {
    size_t i = (size_t)blockIdx.x * 256 + threadIdx.x;
    if (i >= (size_t)512 * 512) return;
    int c = (int)(i % 512);
    int row = (int)(i / 512);
    int t = row & 63;
    float v = 0.f;
    if (t < 60) v = text[(size_t)t * 512 + c] + y[i] * scl[c] + shf[c];
    tdf[i] = v;
    tdbf[i] = f2bf(v);
}

__global__ __launch_bounds__(256) void apply_f1_k(const float* __restrict__ y,
                                                  const float* __restrict__ scl,
                                                  const float* __restrict__ shf,
                                                  u16* __restrict__ o) {
    size_t i = (size_t)blockIdx.x * 256 + threadIdx.x;
    if (i >= (size_t)512 * 1024) return;
    int c = (int)(i % 1024);
    int t = (int)((i / 1024) & 63);
    float v = (t < 60) ? hswish(y[i] * scl[c] + shf[c]) : 0.f;
    o[i] = f2bf(v);
}

__global__ __launch_bounds__(256) void final_out_k(const float* __restrict__ y,
                                                   const float* __restrict__ scl,
                                                   const float* __restrict__ shf,
                                                   const float* __restrict__ tdf,
                                                   float* __restrict__ out) {
    int i = blockIdx.x * 256 + threadIdx.x;
    if (i >= 60 * 512) return;
    int c = i % 512, t = i / 512;                 // batch 0 occupies rows 0..59
    out[i] = y[(size_t)t * 512 + c] * scl[c] + shf[c] + tdf[(size_t)t * 512 + c];
}

// ---------------------------------------------------------------------------
extern "C" void kernel_launch(void* const* d_in, const int* in_sizes, int n_in,
                              void* d_out, int out_size, void* d_ws, size_t ws_size,
                              hipStream_t stream)
{
    (void)in_sizes; (void)n_in; (void)out_size; (void)ws_size;
    const float* im    = (const float*)d_in[0];
    const float* text  = (const float*)d_in[1];
    const float* W_dc  = (const float*)d_in[2];
    const float* g_dc  = (const float*)d_in[3];
    const float* b_dc  = (const float*)d_in[4];
    const float* W_kv  = (const float*)d_in[5];
    const float* g_kv  = (const float*)d_in[6];
    const float* b_kv  = (const float*)d_in[7];
    const float* W_q   = (const float*)d_in[8];
    const float* g_q   = (const float*)d_in[9];
    const float* b_q   = (const float*)d_in[10];
    const float* W_pr  = (const float*)d_in[11];
    const float* g_pr  = (const float*)d_in[12];
    const float* b_pr  = (const float*)d_in[13];
    const float* biases= (const float*)d_in[14];
    const float* W_f1  = (const float*)d_in[15];
    const float* g_f1  = (const float*)d_in[16];
    const float* b_f1  = (const float*)d_in[17];
    const float* W_f2  = (const float*)d_in[18];
    const float* g_f2  = (const float*)d_in[19];
    const float* b_f2  = (const float*)d_in[20];
    float* out = (float*)d_out;
    char*  ws  = (char*)d_ws;

    // ---- workspace layout (phased aliasing; peak ~768 MB) ----
    constexpr size_t O_XBF   = 0;                         // 151MB  (phase 1)
    constexpr size_t O_Y1    = 150994944ull;              // 151MB  (phase 1)
    constexpr size_t O_Y2    = 0;                         // 453MB  (phase 2)
    constexpr size_t O_LOG   = 0;                         // 151MB  (phase 3)
    constexpr size_t O_PRB   = 150994944ull;              // 75.5MB (phase 3)
    constexpr size_t O_OUTB  = O_PRB + 75497472ull;       // 2MB    (phase 3)
    constexpr size_t B1      = 452984832ull;              // persistent region
    constexpr size_t O_XD    = B1;                        // 75.5MB
    constexpr size_t O_KBUF  = B1 + 75497472ull;          // 75.5MB
    constexpr size_t O_VT    = B1 + 150994944ull;         // 151MB
    constexpr size_t B2      = B1 + 301989888ull;         // smalls
    constexpr size_t O_WDC   = B2;
    constexpr size_t O_WKV   = O_WDC  + 1048576ull;
    constexpr size_t O_WPR   = O_WKV  + 1572864ull;
    constexpr size_t O_WF1   = O_WPR  + 1048576ull;
    constexpr size_t O_WF2   = O_WF1  + 1048576ull;
    constexpr size_t O_APROJ = O_WF2  + 1048576ull;
    constexpr size_t O_YPROJ = O_APROJ+ 1048576ull;
    constexpr size_t O_TDF   = O_YPROJ+ 1048576ull;
    constexpr size_t O_TDBF  = O_TDF  + 1048576ull;
    constexpr size_t O_YF1   = O_TDBF + 524288ull;
    constexpr size_t O_F1BF  = O_YF1  + 2097152ull;
    constexpr size_t O_YF2   = O_F1BF + 1048576ull;
    constexpr size_t O_YQ    = O_YF2  + 1048576ull;
    constexpr size_t O_QB    = O_YQ   + 131072ull;
    constexpr size_t O_SUM   = O_QB   + 65536ull;
    constexpr size_t O_SSQ   = O_SUM  + 8192ull;
    constexpr size_t O_SCL   = O_SSQ  + 8192ull;
    constexpr size_t O_SHF   = O_SCL  + 8192ull;

    u16*   xbf   = (u16*)  (ws + O_XBF);
    float* y1    = (float*)(ws + O_Y1);
    float* y2    = (float*)(ws + O_Y2);
    float* logit = (float*)(ws + O_LOG);
    u16*   probs = (u16*)  (ws + O_PRB);
    float* outb  = (float*)(ws + O_OUTB);
    u16*   xd    = (u16*)  (ws + O_XD);
    u16*   kbuf  = (u16*)  (ws + O_KBUF);
    u16*   vT    = (u16*)  (ws + O_VT);
    u16*   wdc   = (u16*)  (ws + O_WDC);
    u16*   wkv   = (u16*)  (ws + O_WKV);
    u16*   wpr   = (u16*)  (ws + O_WPR);
    u16*   wf1   = (u16*)  (ws + O_WF1);
    u16*   wf2   = (u16*)  (ws + O_WF2);
    u16*   aproj = (u16*)  (ws + O_APROJ);
    float* yproj = (float*)(ws + O_YPROJ);
    float* tdf   = (float*)(ws + O_TDF);
    u16*   tdbf  = (u16*)  (ws + O_TDBF);
    float* yf1   = (float*)(ws + O_YF1);
    u16*   f1bf  = (u16*)  (ws + O_F1BF);
    float* yf2   = (float*)(ws + O_YF2);
    float* yq    = (float*)(ws + O_YQ);
    u16*   qb    = (u16*)  (ws + O_QB);
    float* sum   = (float*)(ws + O_SUM);
    float* ssq   = (float*)(ws + O_SSQ);
    float* scl   = (float*)(ws + O_SCL);
    float* shf   = (float*)(ws + O_SHF);

    auto run_stats = [&](const float* Y, int R, int Cc, int period, int valid,
                         float invR, const float* g, const float* bb, int chunks) {
        hipMemsetAsync(ws + O_SUM, 0, 16384, stream);     // sum + ssq
        int rpc = (R + chunks - 1) / chunks;
        dim3 gs((Cc + 255) / 256, chunks);
        colstats<<<gs, 256, 0, stream>>>(Y, sum, ssq, R, Cc, period, valid, rpc);
        make_scaleshift<<<(Cc + 255) / 256, 256, 0, stream>>>(sum, ssq, g, bb, scl, shf, Cc, invR);
    };

    // 0) weights -> bf16
    to_bf16_k<<<(524288 + 255) / 256, 256, 0, stream>>>(W_dc, wdc, 524288);
    to_bf16_k<<<(786432 + 255) / 256, 256, 0, stream>>>(W_kv, wkv, 786432);
    to_bf16_k<<<(524288 + 255) / 256, 256, 0, stream>>>(W_pr, wpr, 524288);
    to_bf16_k<<<(524288 + 255) / 256, 256, 0, stream>>>(W_f1, wf1, 524288);
    to_bf16_k<<<(524288 + 255) / 256, 256, 0, stream>>>(W_f2, wf2, 524288);

    // 1) x = transpose(im) -> bf16 ; y1 = x . W_dc^T ; BN stats ; xd = hswish(BN)
    transpose_to_bf16<<<dim3(288, 32, 8), 256, 0, stream>>>(im, xbf);
    gemm_bf16_nt<<<dim3(512 / 64, 73728 / 128), 256, 0, stream>>>(xbf, wdc, y1, 73728, 512, 1024);
    run_stats(y1, 73728, 512, 64, 64, 1.0f / 73728.f, g_dc, b_dc, 64);
    {
        size_t tot = (size_t)73728 * 512;
        bn_hswish_bf16<<<(unsigned)((tot + 255) / 256), 256, 0, stream>>>(y1, scl, shf, xd, tot, 512);
    }

    // 2) y2 = xd . W_kv^T ; BN ; scatter -> k (B,NH,N,KD), vT (B,NH,D_V,N)
    gemm_bf16_nt<<<dim3(1536 / 64, 73728 / 128), 256, 0, stream>>>(xd, wkv, y2, 73728, 1536, 512);
    run_stats(y2, 73728, 1536, 64, 64, 1.0f / 73728.f, g_kv, b_kv, 64);
    {
        size_t tot = (size_t)73728 * 1536;
        scatter_kv<<<(unsigned)((tot + 255) / 256), 256, 0, stream>>>(y2, scl, shf, kbuf, vT);
    }

    // 3) q (60x512) + BN over 60 rows -> padded 64x512 bf16
    q_gemm<<<60, 512, 0, stream>>>(text, W_q, yq);
    q_bn_fill<<<2, 256, 0, stream>>>(yq, g_q, b_q, qb);

    // 4) attention
    attn_logits<<<dim3(9216 / 16, 64), 128, 0, stream>>>(qb, kbuf, biases, logit);
    softmax_row<<<64 * 64, 256, 0, stream>>>(logit, probs);
    hipMemsetAsync(outb, 0, (size_t)512 * 1024 * 4, stream);
    attn_out<<<dim3(64, 4), 256, 0, stream>>>(probs, vT, outb);

    // 5) proj: BN(hswish(out) . W_proj^T) + text -> textd
    hswish_to_bf16<<<(524288 + 255) / 256, 256, 0, stream>>>(outb, aproj, 524288);
    gemm_bf16_nt<<<dim3(512 / 64, 512 / 128), 256, 0, stream>>>(aproj, wpr, yproj, 512, 512, 1024);
    run_stats(yproj, 512, 512, 64, 60, 1.0f / 480.f, g_pr, b_pr, 4);
    apply_proj_k<<<(512 * 512 + 255) / 256, 256, 0, stream>>>(yproj, scl, shf, text, tdf, tdbf);

    // 6) FFN: f = BN(hswish(BN(textd.Wf1^T)).Wf2^T) ; out = f + textd (batch 0)
    gemm_bf16_nt<<<dim3(1024 / 64, 512 / 128), 256, 0, stream>>>(tdbf, wf1, yf1, 512, 1024, 512);
    run_stats(yf1, 512, 1024, 64, 60, 1.0f / 480.f, g_f1, b_f1, 4);
    apply_f1_k<<<(512 * 1024 + 255) / 256, 256, 0, stream>>>(yf1, scl, shf, f1bf);
    gemm_bf16_nt<<<dim3(512 / 64, 512 / 128), 256, 0, stream>>>(f1bf, wf2, yf2, 512, 512, 1024);
    run_stats(yf2, 512, 512, 64, 60, 1.0f / 480.f, g_f2, b_f2, 4);
    final_out_k<<<(60 * 512 + 255) / 256, 256, 0, stream>>>(yf2, scl, shf, tdf, out);
}
